// GraphAttentionLayer_58841051955384
// MI455X (gfx1250) — matrix operand order repaired and checked by reading, not compiled, not run on hardware
//
#include <hip/hip_runtime.h>

typedef __bf16 bf16x16 __attribute__((ext_vector_type(16)));
typedef float  f32x8   __attribute__((ext_vector_type(8)));

static constexpr int   B_      = 8;
static constexpr int   NT      = 2048;   // tokens per batch
static constexpr int   FD      = 256;    // feature dim (in == out)
static constexpr float ALPHA_  = 0.2f;
static constexpr float NEG_BIG = -9.0e15f;

union FragU { uint4 q[2]; bf16x16 v; };

__device__ __forceinline__ unsigned short f32_to_bf16(float f) {
  unsigned u = __builtin_bit_cast(unsigned, f);
  u += 0x7FFFu + ((u >> 16) & 1u);           // round-to-nearest-even
  return (unsigned short)(u >> 16);
}
__device__ __forceinline__ float bf16_to_f32(unsigned short h) {
  unsigned u = ((unsigned)h) << 16;
  return __builtin_bit_cast(float, u);
}
__device__ __forceinline__ float lrelu(float e) { return e > 0.0f ? e : ALPHA_ * e; }

__device__ __forceinline__ f32x8 wmma_bf16(const FragU& a, const FragU& b, f32x8 c) {
  return __builtin_amdgcn_wmma_f32_16x16x32_bf16(false, a.v, false, b.v,
                                                 (short)0, c, false, false);
}

// ---------------------------------------------------------------------------
// Kernel 1: Wh = x @ W  (bf16x3 split for ~fp32 accuracy, f32 accumulate)
// Block tile 128(M) x 128(N), 8 waves as 4(M) x 2(N), wave tile 32x64.
// Writes Wh (f32, row-major) and WhT (bf16, [b][col][token] for GEMM2 B-frags).
// ---------------------------------------------------------------------------
static constexpr int BM = 128, BN = 128, BK = 32;
static constexpr int LDA = 40;  // u16 stride (pad: 16B aligned + bank spread)
static constexpr int LDB = 40;

__global__ __launch_bounds__(256) void k_gemm_xw(
    const float* __restrict__ x, const float* __restrict__ W,
    float* __restrict__ Wh, unsigned short* __restrict__ WhT)
{
  __shared__ unsigned short Ahi[BM * LDA], Alo[BM * LDA];
  __shared__ unsigned short Bhi[BN * LDB], Blo[BN * LDB];

  const int b    = blockIdx.z;
  const int mblk = blockIdx.x * BM;
  const int nblk = blockIdx.y * BN;
  const int tid  = threadIdx.x;
  const int lane = tid & 31, wave = tid >> 5;
  const int wr   = wave & 3, wc = wave >> 2;     // wave grid 4 x 2
  const int l    = lane & 15, h = lane >> 4;

  f32x8 acc[2][4] = {};
  const float* xb = x + (size_t)b * NT * FD;

  for (int k0 = 0; k0 < FD; k0 += BK) {
    // Stage A: x[mblk..+127][k0..+31] f32 -> hi/lo bf16 (row-major)
    #pragma unroll
    for (int e = 0; e < 16; ++e) {
      int id = tid + 256 * e;                 // 4096 elements
      int r = id >> 5, k = id & 31;
      float v = xb[(size_t)(mblk + r) * FD + k0 + k];
      unsigned short hs = f32_to_bf16(v);
      Ahi[r * LDA + k] = hs;
      Alo[r * LDA + k] = f32_to_bf16(v - bf16_to_f32(hs));
    }
    // Stage W^T: W[k0..+31][nblk..+127] -> hi/lo bf16, transposed [col][k]
    #pragma unroll
    for (int e = 0; e < 16; ++e) {
      int id = tid + 256 * e;
      int k = id >> 7, c = id & 127;
      float v = W[(size_t)(k0 + k) * FD + nblk + c];
      unsigned short hs = f32_to_bf16(v);
      Bhi[c * LDB + k] = hs;
      Blo[c * LDB + k] = f32_to_bf16(v - bf16_to_f32(hs));
    }
    __syncthreads();

    FragU a_hi[2], a_lo[2], b_hi[4], b_lo[4];
    #pragma unroll
    for (int mt = 0; mt < 2; ++mt) {
      int row = wr * 32 + mt * 16 + l;
      a_hi[mt].q[0] = *(const uint4*)&Ahi[row * LDA + 16 * h];
      a_hi[mt].q[1] = *(const uint4*)&Ahi[row * LDA + 16 * h + 8];
      a_lo[mt].q[0] = *(const uint4*)&Alo[row * LDA + 16 * h];
      a_lo[mt].q[1] = *(const uint4*)&Alo[row * LDA + 16 * h + 8];
    }
    #pragma unroll
    for (int nt = 0; nt < 4; ++nt) {
      int col = wc * 64 + nt * 16 + l;
      b_hi[nt].q[0] = *(const uint4*)&Bhi[col * LDB + 16 * h];
      b_hi[nt].q[1] = *(const uint4*)&Bhi[col * LDB + 16 * h + 8];
      b_lo[nt].q[0] = *(const uint4*)&Blo[col * LDB + 16 * h];
      b_lo[nt].q[1] = *(const uint4*)&Blo[col * LDB + 16 * h + 8];
    }
    #pragma unroll
    for (int mt = 0; mt < 2; ++mt)
      #pragma unroll
      for (int nt = 0; nt < 4; ++nt) {
        acc[mt][nt] = wmma_bf16(a_hi[mt], b_hi[nt], acc[mt][nt]);
        acc[mt][nt] = wmma_bf16(a_hi[mt], b_lo[nt], acc[mt][nt]);
        acc[mt][nt] = wmma_bf16(a_lo[mt], b_hi[nt], acc[mt][nt]);
      }
    __syncthreads();
  }

  // Epilogue: Wh f32 + transposed bf16 copy
  #pragma unroll
  for (int mt = 0; mt < 2; ++mt)
    #pragma unroll
    for (int nt = 0; nt < 4; ++nt)
      #pragma unroll
      for (int v = 0; v < 8; ++v) {
        int row_g = mblk + wr * 32 + mt * 16 + v + 8 * h;
        int col_g = nblk + wc * 64 + nt * 16 + l;
        float val = acc[mt][nt][v];
        Wh [((size_t)b * NT + row_g) * FD + col_g] = val;
        WhT[((size_t)b * FD + col_g) * NT + row_g] = f32_to_bf16(val);
      }
}

// ---------------------------------------------------------------------------
// Kernel 2: f1 = Wh @ a1, f2 = Wh @ a2   (one wave per token row)
// ---------------------------------------------------------------------------
__global__ __launch_bounds__(256) void k_rowdot(
    const float* __restrict__ Wh, const float* __restrict__ a,
    float* __restrict__ f1, float* __restrict__ f2)
{
  int b = blockIdx.y;
  int r = blockIdx.x * 8 + (threadIdx.x >> 5);
  int lane = threadIdx.x & 31;
  const float* row = Wh + ((size_t)b * NT + r) * FD;
  float s1 = 0.0f, s2 = 0.0f;
  #pragma unroll
  for (int q = 0; q < 8; ++q) {
    int idx = q * 32 + lane;
    float w = row[idx];
    s1 += w * a[idx];
    s2 += w * a[FD + idx];
  }
  #pragma unroll
  for (int off = 16; off >= 1; off >>= 1) {
    s1 += __shfl_xor(s1, off, 32);
    s2 += __shfl_xor(s2, off, 32);
  }
  if (lane == 0) { f1[b * NT + r] = s1; f2[b * NT + r] = s2; }
}

// ---------------------------------------------------------------------------
// Kernel 3: per-row online softmax stats (max m, 1/sum).  One wave per row.
// NEG_BIG is finite, so the all-masked row needs no special case.
// ---------------------------------------------------------------------------
__global__ __launch_bounds__(256) void k_stats(
    const int* __restrict__ adj, const float* __restrict__ f1,
    const float* __restrict__ f2, float* __restrict__ mrow,
    float* __restrict__ rsrow)
{
  int b = blockIdx.y;
  int r = blockIdx.x * 8 + (threadIdx.x >> 5);
  int lane = threadIdx.x & 31;
  const int*   ar  = adj + ((size_t)b * NT + r) * NT;
  const float* f2b = f2 + b * NT;
  float f1i = f1[b * NT + r];
  float m = -3.0e38f, s = 0.0f;
  for (int q = 0; q < NT / 32; ++q) {
    int j = q * 32 + lane;
    float e = ar[j] > 0 ? lrelu(f1i + f2b[j]) : NEG_BIG;
    float nm = fmaxf(m, e);
    s = s * __expf(m - nm) + __expf(e - nm);
    m = nm;
  }
  #pragma unroll
  for (int off = 16; off >= 1; off >>= 1) {
    float mo = __shfl_xor(m, off, 32);
    float so = __shfl_xor(s, off, 32);
    float nm = fmaxf(m, mo);
    s = s * __expf(m - nm) + so * __expf(mo - nm);
    m = nm;
  }
  if (lane == 0) { mrow[b * NT + r] = m; rsrow[b * NT + r] = 1.0f / s; }
}

// ---------------------------------------------------------------------------
// Kernel 4: h' = elu( softmax(masked e) @ Wh ), flash-style.
// Block: batch b, 16 rows; 8 waves x 32 cols = full 256 cols.
// P tile (16x32 bf16) built cooperatively in double-buffered LDS (1 barrier).
// ---------------------------------------------------------------------------
static constexpr int LDP = 40;   // u16 stride for P rows

__global__ __launch_bounds__(256) void k_attn(
    const int* __restrict__ adj, const unsigned short* __restrict__ WhT,
    const float* __restrict__ f1, const float* __restrict__ f2,
    const float* __restrict__ mrow, const float* __restrict__ rsrow,
    float* __restrict__ out)
{
  __shared__ unsigned short P[2][16 * LDP];

  const int b  = blockIdx.y;
  const int i0 = blockIdx.x * 16;
  const int tid = threadIdx.x;
  const int lane = tid & 31, wave = tid >> 5;
  const int l = lane & 15, h = lane >> 4;
  const int lrow = tid >> 4, kk = (tid & 15) * 2;   // cooperative P mapping

  const float f1r = f1[b * NT + i0 + lrow];
  const float mr  = mrow[b * NT + i0 + lrow];
  const float rs  = rsrow[b * NT + i0 + lrow];
  const int*   adjr = adj + ((size_t)b * NT + i0 + lrow) * NT;
  const float* f2b  = f2 + b * NT;
  const unsigned short* WhTb = WhT + (size_t)b * FD * NT;

  f32x8 acc[2] = {};

  int ph = 0;
  for (int j0 = 0; j0 < NT; j0 += 32, ph ^= 1) {
    // Build P[l][k] = exp(masked_e - m) / s  (2 elements per thread)
    int2 av = *(const int2*)&adjr[j0 + kk];
    float e0 = av.x > 0 ? lrelu(f1r + f2b[j0 + kk])     : NEG_BIG;
    float e1 = av.y > 0 ? lrelu(f1r + f2b[j0 + kk + 1]) : NEG_BIG;
    P[ph][lrow * LDP + kk]     = f32_to_bf16(__expf(e0 - mr) * rs);
    P[ph][lrow * LDP + kk + 1] = f32_to_bf16(__expf(e1 - mr) * rs);
    if (j0 + 32 < NT) __builtin_prefetch(&adjr[j0 + 32 + kk], 0, 1);
    __syncthreads();

    // A fragment from LDS P tile
    FragU af;
    af.q[0] = *(const uint4*)&P[ph][l * LDP + 16 * h];
    af.q[1] = *(const uint4*)&P[ph][l * LDP + 16 * h + 8];

    // B fragments from transposed Wh (contiguous 32B per lane)
    #pragma unroll
    for (int nt = 0; nt < 2; ++nt) {
      int col = wave * 32 + nt * 16 + l;
      const unsigned short* src = &WhTb[(size_t)col * NT + j0 + 16 * h];
      FragU bf;
      bf.q[0] = *(const uint4*)&src[0];
      bf.q[1] = *(const uint4*)&src[8];
      acc[nt] = wmma_bf16(af, bf, acc[nt]);
    }
    // no trailing barrier needed: next iter writes the other buffer
  }

  // Fused ELU epilogue
  #pragma unroll
  for (int nt = 0; nt < 2; ++nt)
    #pragma unroll
    for (int v = 0; v < 8; ++v) {
      int row = i0 + v + 8 * h;
      int col = wave * 32 + nt * 16 + l;
      float x = acc[nt][v];
      out[((size_t)b * NT + row) * FD + col] = x > 0.0f ? x : __expf(x) - 1.0f;
    }
}

// ---------------------------------------------------------------------------
extern "C" void kernel_launch(void* const* d_in, const int* in_sizes, int n_in,
                              void* d_out, int out_size, void* d_ws, size_t ws_size,
                              hipStream_t stream) {
  const float* x   = (const float*)d_in[0];   // (8, 2048, 256)
  const int*   adj = (const int*)  d_in[1];   // (8, 2048, 2048)
  const float* W   = (const float*)d_in[2];   // (256, 256)
  const float* a   = (const float*)d_in[3];   // (512, 1)
  float* out = (float*)d_out;                 // (8, 2048, 256)

  char* ws = (char*)d_ws;
  const size_t whF32Bytes = (size_t)B_ * NT * FD * sizeof(float);      // 16.78 MB
  const size_t whBfBytes  = (size_t)B_ * NT * FD * sizeof(short);      //  8.39 MB
  float*          Wh  = (float*)ws;                        ws += whF32Bytes;
  unsigned short* WhT = (unsigned short*)ws;               ws += whBfBytes;
  float* f1  = (float*)ws;  ws += (size_t)B_ * NT * sizeof(float);
  float* f2  = (float*)ws;  ws += (size_t)B_ * NT * sizeof(float);
  float* mr  = (float*)ws;  ws += (size_t)B_ * NT * sizeof(float);
  float* rsr = (float*)ws;

  k_gemm_xw<<<dim3(NT / BM, FD / BN, B_), 256, 0, stream>>>(x, W, Wh, WhT);
  k_rowdot <<<dim3(NT / 8, B_),           256, 0, stream>>>(Wh, a, f1, f2);
  k_stats  <<<dim3(NT / 8, B_),           256, 0, stream>>>(adj, f1, f2, mr, rsr);
  k_attn   <<<dim3(NT / 16, B_),          256, 0, stream>>>(adj, WhT, f1, f2, mr, rsr, out);
}
